// FastLinearAttentionLayer_12120397710193
// MI455X (gfx1250) — compile-verified, matrix operand order checked by reading, not run on hardware
//
#include <hip/hip_runtime.h>
#include <hip/hip_bf16.h>

typedef __bf16 bf16_t;
typedef __attribute__((ext_vector_type(16))) __bf16 bf16x16;
typedef __attribute__((ext_vector_type(8)))  __bf16 bf16x8;
typedef __attribute__((ext_vector_type(8)))  float  f32x8;

#define N_B  4
#define N_S  8192
#define N_E  512
#define N_F  2048
#define N_H  8
#define N_DH 64
#define N_M  (N_B * N_S)   // 32768 tokens
#define N_BH (N_B * N_H)   // 32 head-batches

// ---------------------------------------------------------------------------
// CDNA5 primitives
// ---------------------------------------------------------------------------
__device__ __forceinline__ f32x8 wmma_bf16(bf16x16 a, bf16x16 b, f32x8 c) {
  return __builtin_amdgcn_wmma_f32_16x16x32_bf16(false, a, false, b, (short)0, c,
                                                 false, false);
}

__device__ __forceinline__ f32x8 zero8() {
  f32x8 z = {0.f, 0.f, 0.f, 0.f, 0.f, 0.f, 0.f, 0.f};
  return z;
}

// Async copy 16 bytes/lane from global directly into LDS (ASYNCcnt-tracked).
// Generic __shared__ pointer low 32 bits == LDS byte offset (aperture rule).
__device__ __forceinline__ void async_cp_b128(void* lds, const void* g) {
  unsigned l = (unsigned)(size_t)lds;
  asm volatile("global_load_async_to_lds_b128 %0, %1, off"
               :: "v"(l), "v"(g) : "memory");
}

__device__ __forceinline__ void wait_async_3() {
  asm volatile("s_wait_asynccnt 0x3" ::: "memory");
}
__device__ __forceinline__ void wait_async_0() {
  asm volatile("s_wait_asynccnt 0x0" ::: "memory");
}

// Gather one 16x32 bf16 fragment from LDS (row-major for A, N-major for B).
// ISA layout: lane -> m = lane%16; K = (lane/16)*8 + i (i<8), +16 for i>=8.
__device__ __forceinline__ bf16x16 frag_ld(const bf16_t* base, int row0, int col0,
                                           int ld) {
  const int lane = threadIdx.x & 31;
  const bf16_t* p = base + (row0 + (lane & 15)) * ld + col0 + ((lane >> 4) << 3);
  bf16x8 lo = *(const bf16x8*)p;
  bf16x8 hi = *(const bf16x8*)(p + 16);
  bf16x16 r;
#pragma unroll
  for (int i = 0; i < 8; ++i) { r[i] = lo[i]; r[8 + i] = hi[i]; }
  return r;
}

// ---------------------------------------------------------------------------
// Elementwise f32 -> bf16 convert
// ---------------------------------------------------------------------------
__global__ __launch_bounds__(256) void k_cvt(const float* __restrict__ x,
                                             bf16_t* __restrict__ y, int n) {
  int i = blockIdx.x * 1024 + threadIdx.x;
#pragma unroll
  for (int j = 0; j < 4; ++j, i += 256)
    if (i < n) y[i] = (bf16_t)x[i];
}

// f32 [K,N] -> bf16 [N,K] transpose (32x32 LDS tiles, 256 threads)
__global__ __launch_bounds__(256) void k_cvtT(const float* __restrict__ W,
                                              bf16_t* __restrict__ Wt, int K,
                                              int N) {
  __shared__ float tile[32][33];
  const int k0 = blockIdx.x * 32, n0 = blockIdx.y * 32;
  const int t = threadIdx.x, tx = t & 31, ty = t >> 5;
#pragma unroll
  for (int i = 0; i < 4; ++i)
    tile[ty + i * 8][tx] = W[(size_t)(k0 + ty + i * 8) * N + n0 + tx];
  __syncthreads();
#pragma unroll
  for (int i = 0; i < 4; ++i)
    Wt[(size_t)(n0 + ty + i * 8) * K + k0 + tx] = (bf16_t)tile[tx][ty + i * 8];
}

// ---------------------------------------------------------------------------
// Unified bf16 GEMM:  C = A[M,K](bf16) @ Bt[N,K]^T(bf16) + bias.
// Double-buffered LDS filled by global_load_async_to_lds_b128.
// act: 0 none, 1 relu, 2 elu+1.  headLayout: write outB as [BH,S,Dh].
// Block tile 128x64, 8 waves of 32x32 (2x2 wmma), K-chunks of 32.
// ---------------------------------------------------------------------------
__global__ __launch_bounds__(256) void k_gemm(
    const bf16_t* __restrict__ A, const bf16_t* __restrict__ Bt,
    const float* __restrict__ bias, const float* __restrict__ resid,
    float* __restrict__ outF, bf16_t* __restrict__ outB, int K, int N, int act,
    int headLayout) {
  __shared__ __align__(16) bf16_t As[2][128 * 32];
  __shared__ __align__(16) bf16_t Bs[2][64 * 32];
  const int t = threadIdx.x;
  const int wave = t >> 5, lane = t & 31;
  const int m0 = blockIdx.x * 128, n0 = blockIdx.y * 64;
  const int mw = (wave & 3) * 32, nw = (wave >> 2) * 32;

  // Per-chunk async stage: 3 b128 instructions per wave.
  auto stage = [&](int kc, int buf) {
    {  // A tile: 128 rows x 64B = 512 units, units t and t+256
      int row = t >> 2, c = (t & 3) << 3;
      async_cp_b128(&As[buf][row * 32 + c],
                    A + (size_t)(m0 + row) * K + kc * 32 + c);
      int u2 = t + 256;
      int row2 = u2 >> 2, c2 = (u2 & 3) << 3;
      async_cp_b128(&As[buf][row2 * 32 + c2],
                    A + (size_t)(m0 + row2) * K + kc * 32 + c2);
    }
    {  // B tile: 64 rows x 64B = 256 units
      int row = t >> 2, c = (t & 3) << 3;
      async_cp_b128(&Bs[buf][row * 32 + c],
                    Bt + (size_t)(n0 + row) * K + kc * 32 + c);
    }
  };

  f32x8 acc[2][2];
#pragma unroll
  for (int i = 0; i < 2; ++i)
#pragma unroll
    for (int j = 0; j < 2; ++j) acc[i][j] = zero8();

  const int nk = K / 32;
  stage(0, 0);
  for (int kc = 0; kc < nk; ++kc) {
    const int cur = kc & 1;
    const bool more = (kc + 1) < nk;
    if (more) stage(kc + 1, cur ^ 1);   // overwrites buffer last read at kc-1,
                                        // protected by the trailing barrier
    if (more) wait_async_3();           // our chunk-kc copies done
    else      wait_async_0();
    __syncthreads();                    // everyone's chunk-kc copies done
    bf16x16 a0 = frag_ld(As[cur], mw, 0, 32);
    bf16x16 a1 = frag_ld(As[cur], mw + 16, 0, 32);
    bf16x16 b0 = frag_ld(Bs[cur], nw, 0, 32);
    bf16x16 b1 = frag_ld(Bs[cur], nw + 16, 0, 32);
    acc[0][0] = wmma_bf16(a0, b0, acc[0][0]);
    acc[0][1] = wmma_bf16(a0, b1, acc[0][1]);
    acc[1][0] = wmma_bf16(a1, b0, acc[1][0]);
    acc[1][1] = wmma_bf16(a1, b1, acc[1][1]);
    __syncthreads();                    // done reading buf[cur]
  }

  const int lm = lane & 15, lh = lane >> 4;
#pragma unroll
  for (int i = 0; i < 2; ++i)
#pragma unroll
    for (int j = 0; j < 2; ++j) {
      int gn = n0 + nw + j * 16 + lm;
      float bb = bias[gn];
#pragma unroll
      for (int r = 0; r < 8; ++r) {
        int gm = m0 + mw + i * 16 + lh * 8 + r;
        float v = acc[i][j][r] + bb;
        if (act == 1) v = fmaxf(v, 0.f);
        if (act == 2) v = v > 0.f ? v + 1.f : __expf(v);  // elu+1
        if (headLayout) {
          int b = gm >> 13, s = gm & (N_S - 1);
          int h = gn >> 6, d = gn & 63;
          outB[((size_t)(b * N_H + h) * N_S + s) * N_DH + d] = (bf16_t)v;
        } else {
          size_t off = (size_t)gm * N + gn;
          if (resid) v += resid[off];
          if (outF) outF[off] = v;
          if (outB) outB[off] = (bf16_t)v;
        }
      }
    }
}

// ---------------------------------------------------------------------------
// kv[bh] += km^T @ v over an S-chunk of 512, plus ksum.  Grid (32, 16).
// ---------------------------------------------------------------------------
__global__ __launch_bounds__(256) void k_kv(const bf16_t* __restrict__ km,
                                            const bf16_t* __restrict__ vv,
                                            float* __restrict__ kv,
                                            float* __restrict__ ksum) {
  __shared__ __align__(16) bf16_t kmT[64 * 32];  // [d][s]
  __shared__ __align__(16) bf16_t vT[64 * 32];   // [e][s]
  const int bh = blockIdx.x;
  const int s0 = blockIdx.y * 512;
  const int t = threadIdx.x, wave = t >> 5, lane = t & 31;
  const bf16_t* kmp = km + (size_t)bh * N_S * N_DH;
  const bf16_t* vp = vv + (size_t)bh * N_S * N_DH;

  const int mw = (wave & 3) * 16;
  const int nw = (wave >> 2) * 32;
  f32x8 acc0 = zero8(), acc1 = zero8();

  for (int kc = 0; kc < 16; ++kc) {
    int sb = s0 + kc * 32;
#pragma unroll
    for (int i = 0; i < 8; ++i) {
      int idx = i * 256 + t;
      int ss = idx >> 6, d = idx & 63;
      kmT[d * 32 + ss] = kmp[(size_t)(sb + ss) * N_DH + d];
      vT[d * 32 + ss] = vp[(size_t)(sb + ss) * N_DH + d];
    }
    __syncthreads();
    bf16x16 a = frag_ld(kmT, mw, 0, 32);
    bf16x16 b0 = frag_ld(vT, nw, 0, 32);
    bf16x16 b1 = frag_ld(vT, nw + 16, 0, 32);
    acc0 = wmma_bf16(a, b0, acc0);
    acc1 = wmma_bf16(a, b1, acc1);
    __syncthreads();
  }

  float* kvp = kv + (size_t)bh * N_DH * N_DH;
  const int lm = lane & 15, lh = lane >> 4;
#pragma unroll
  for (int r = 0; r < 8; ++r) {
    int d = mw + lh * 8 + r;
    atomicAdd(&kvp[d * N_DH + nw + lm], acc0[r]);
    atomicAdd(&kvp[d * N_DH + nw + 16 + lm], acc1[r]);
  }
  int d = t & 63, g = t >> 6;
  float ps = 0.f;
  for (int ss = g; ss < 512; ss += 4)
    ps += (float)kmp[(size_t)(s0 + ss) * N_DH + d];
  atomicAdd(&ksum[bh * N_DH + d], ps);
}

// ---------------------------------------------------------------------------
// attn = (qm @ kv) * z,  z = 1/(qm . ksum + 1e-6).  Grid (32, 64).
// ---------------------------------------------------------------------------
__global__ __launch_bounds__(256) void k_attn(const bf16_t* __restrict__ qm,
                                              const float* __restrict__ kv,
                                              const float* __restrict__ ksum,
                                              bf16_t* __restrict__ attnb) {
  __shared__ __align__(16) bf16_t qs[128 * 64];  // [row][d]
  __shared__ __align__(16) bf16_t kvT[64 * 64];  // [e][d]
  __shared__ float zs[128];
  __shared__ float ks[64];
  const int bh = blockIdx.x;
  const int s0 = blockIdx.y * 128;
  const int t = threadIdx.x, wave = t >> 5, lane = t & 31;
  const bf16_t* qp = qm + (size_t)bh * N_S * N_DH;
  const float* kvp = kv + (size_t)bh * N_DH * N_DH;

  if (t < 64) ks[t] = ksum[bh * N_DH + t];
#pragma unroll
  for (int i = 0; i < 16; ++i) {
    int idx = i * 256 + t;
    int d = idx >> 6, e = idx & 63;
    kvT[e * 64 + d] = (bf16_t)kvp[d * N_DH + e];
  }
#pragma unroll
  for (int i = 0; i < 32; ++i) {
    int idx = i * 256 + t;
    int r = idx >> 6, d = idx & 63;
    qs[r * 64 + d] = qp[(size_t)(s0 + r) * N_DH + d];
  }
  __syncthreads();
  if (t < 128) {
    float dot = 0.f;
#pragma unroll 8
    for (int d = 0; d < 64; ++d) dot += (float)qs[t * 64 + d] * ks[d];
    zs[t] = 1.f / (dot + 1e-6f);
  }
  __syncthreads();

  const int m0 = wave * 16;
  f32x8 acc[4];
#pragma unroll
  for (int nt = 0; nt < 4; ++nt) acc[nt] = zero8();
#pragma unroll
  for (int kc = 0; kc < 2; ++kc) {
    bf16x16 a = frag_ld(qs, m0, kc * 32, 64);
#pragma unroll
    for (int nt = 0; nt < 4; ++nt)
      acc[nt] = wmma_bf16(a, frag_ld(kvT, nt * 16, kc * 32, 64), acc[nt]);
  }

  const int b = bh >> 3, h = bh & 7;
  const int lm = lane & 15, lh = lane >> 4;
#pragma unroll
  for (int nt = 0; nt < 4; ++nt)
#pragma unroll
    for (int r = 0; r < 8; ++r) {
      int row = m0 + lh * 8 + r;
      float v = acc[nt][r] * zs[row];
      attnb[((size_t)(b * N_S + s0 + row)) * N_E + h * N_DH + nt * 16 + lm] =
          (bf16_t)v;
    }
}

// ---------------------------------------------------------------------------
// Rowwise LayerNorm over E=512, one row per 256-thread block.
// ---------------------------------------------------------------------------
__global__ __launch_bounds__(256) void k_ln(const float* __restrict__ y,
                                            const float* __restrict__ g,
                                            const float* __restrict__ be,
                                            float* __restrict__ outF,
                                            bf16_t* __restrict__ outB) {
  __shared__ float red[256];
  __shared__ float red2[256];
  const int row = blockIdx.x, t = threadIdx.x;
  const float* yr = y + (size_t)row * N_E;
  float a = yr[t], b = yr[t + 256];
  red[t] = a + b;
  red2[t] = a * a + b * b;
  __syncthreads();
  for (int off = 128; off > 0; off >>= 1) {
    if (t < off) {
      red[t] += red[t + off];
      red2[t] += red2[t + off];
    }
    __syncthreads();
  }
  float mean = red[0] * (1.f / N_E);
  float var = red2[0] * (1.f / N_E) - mean * mean;
  float rstd = rsqrtf(var + 1e-5f);
  float o1 = (a - mean) * rstd * g[t] + be[t];
  float o2 = (b - mean) * rstd * g[t + 256] + be[t + 256];
  outF[(size_t)row * N_E + t] = o1;
  outF[(size_t)row * N_E + t + 256] = o2;
  if (outB) {
    outB[(size_t)row * N_E + t] = (bf16_t)o1;
    outB[(size_t)row * N_E + t + 256] = (bf16_t)o2;
  }
}

// ---------------------------------------------------------------------------
// Host launcher
// ---------------------------------------------------------------------------
extern "C" void kernel_launch(void* const* d_in, const int* in_sizes, int n_in,
                              void* d_out, int out_size, void* d_ws,
                              size_t ws_size, hipStream_t stream) {
  (void)in_sizes; (void)n_in; (void)out_size; (void)ws_size;
  const float* src = (const float*)d_in[0];
  const float* Wq = (const float*)d_in[1];
  const float* bq = (const float*)d_in[2];
  const float* Wk = (const float*)d_in[3];
  const float* bk = (const float*)d_in[4];
  const float* Wv = (const float*)d_in[5];
  const float* bv = (const float*)d_in[6];
  const float* Wo = (const float*)d_in[7];
  const float* bo = (const float*)d_in[8];
  const float* W1 = (const float*)d_in[9];
  const float* b1 = (const float*)d_in[10];
  const float* W2 = (const float*)d_in[11];
  const float* b2 = (const float*)d_in[12];
  const float* g1 = (const float*)d_in[13];
  const float* be1 = (const float*)d_in[14];
  const float* g2 = (const float*)d_in[15];
  const float* be2 = (const float*)d_in[16];
  float* out = (float*)d_out;

  char* ws = (char*)d_ws;
  size_t off = 0;
  auto carve = [&](size_t bytes) -> char* {
    char* p = ws + off;
    off = (off + bytes + 255) & ~(size_t)255;
    return p;
  };
  bf16_t* srcb = (bf16_t*)carve((size_t)N_M * N_E * 2);
  bf16_t* wqt = (bf16_t*)carve((size_t)N_E * N_E * 2);
  bf16_t* wkt = (bf16_t*)carve((size_t)N_E * N_E * 2);
  bf16_t* wvt = (bf16_t*)carve((size_t)N_E * N_E * 2);
  bf16_t* wot = (bf16_t*)carve((size_t)N_E * N_E * 2);
  bf16_t* w1t = (bf16_t*)carve((size_t)N_E * N_F * 2);
  bf16_t* w2t = (bf16_t*)carve((size_t)N_F * N_E * 2);
  bf16_t* qm = (bf16_t*)carve((size_t)N_BH * N_S * N_DH * 2);
  bf16_t* km = (bf16_t*)carve((size_t)N_BH * N_S * N_DH * 2);
  bf16_t* vb = (bf16_t*)carve((size_t)N_BH * N_S * N_DH * 2);
  float* kv = (float*)carve((size_t)N_BH * N_DH * N_DH * 4);
  float* ksum = (float*)carve((size_t)N_BH * N_DH * 4);
  bf16_t* attnb = (bf16_t*)carve((size_t)N_M * N_E * 2);
  float* ybuf = (float*)carve((size_t)N_M * N_E * 4);
  float* xf = (float*)carve((size_t)N_M * N_E * 4);
  bf16_t* xb = (bf16_t*)carve((size_t)N_M * N_E * 2);
  bf16_t* hb = (bf16_t*)carve((size_t)N_M * N_F * 2);

  hipMemsetAsync(kv, 0, (size_t)N_BH * N_DH * N_DH * 4, stream);
  hipMemsetAsync(ksum, 0, (size_t)N_BH * N_DH * 4, stream);

  dim3 blk(256);
  // precision/layout prep: bf16 activations, bf16 pre-transposed weights
  hipLaunchKernelGGL(k_cvt, dim3((N_M * N_E) / 1024), blk, 0, stream, src, srcb,
                     N_M * N_E);
  hipLaunchKernelGGL(k_cvtT, dim3(N_E / 32, N_E / 32), blk, 0, stream, Wq, wqt,
                     N_E, N_E);
  hipLaunchKernelGGL(k_cvtT, dim3(N_E / 32, N_E / 32), blk, 0, stream, Wk, wkt,
                     N_E, N_E);
  hipLaunchKernelGGL(k_cvtT, dim3(N_E / 32, N_E / 32), blk, 0, stream, Wv, wvt,
                     N_E, N_E);
  hipLaunchKernelGGL(k_cvtT, dim3(N_E / 32, N_E / 32), blk, 0, stream, Wo, wot,
                     N_E, N_E);
  hipLaunchKernelGGL(k_cvtT, dim3(N_E / 32, N_F / 32), blk, 0, stream, W1, w1t,
                     N_E, N_F);
  hipLaunchKernelGGL(k_cvtT, dim3(N_F / 32, N_E / 32), blk, 0, stream, W2, w2t,
                     N_F, N_E);

  dim3 gE(N_M / 128, N_E / 64);
  // QKV projections (elu+1 on q,k), head-major bf16 outputs
  hipLaunchKernelGGL(k_gemm, gE, blk, 0, stream, srcb, wqt, bq,
                     (const float*)nullptr, (float*)nullptr, qm, N_E, N_E, 2, 1);
  hipLaunchKernelGGL(k_gemm, gE, blk, 0, stream, srcb, wkt, bk,
                     (const float*)nullptr, (float*)nullptr, km, N_E, N_E, 2, 1);
  hipLaunchKernelGGL(k_gemm, gE, blk, 0, stream, srcb, wvt, bv,
                     (const float*)nullptr, (float*)nullptr, vb, N_E, N_E, 0, 1);
  hipLaunchKernelGGL(k_kv, dim3(N_BH, N_S / 512), blk, 0, stream, km, vb, kv,
                     ksum);
  hipLaunchKernelGGL(k_attn, dim3(N_BH, N_S / 128), blk, 0, stream, qm, kv,
                     ksum, attnb);
  // out-projection + residual, then LN1
  hipLaunchKernelGGL(k_gemm, gE, blk, 0, stream, attnb, wot, bo, src, ybuf,
                     (bf16_t*)nullptr, N_E, N_E, 0, 0);
  hipLaunchKernelGGL(k_ln, dim3(N_M), blk, 0, stream, ybuf, g1, be1, xf, xb);
  // FFN
  hipLaunchKernelGGL(k_gemm, dim3(N_M / 128, N_F / 64), blk, 0, stream, xb, w1t,
                     b1, (const float*)nullptr, (float*)nullptr, hb, N_E, N_F,
                     1, 0);
  hipLaunchKernelGGL(k_gemm, gE, blk, 0, stream, hb, w2t, b2, xf, ybuf,
                     (bf16_t*)nullptr, N_F, N_E, 0, 0);
  hipLaunchKernelGGL(k_ln, dim3(N_M), blk, 0, stream, ybuf, g2, be2, out,
                     (bf16_t*)nullptr);
}
